// ContextAwareContrastiveEmbeddingGenerator_18202071401103
// MI455X (gfx1250) — compile-verified, matrix operand order checked by reading, not run on hardware
//
#include <hip/hip_runtime.h>
#include <hip/hip_bf16.h>

typedef __attribute__((ext_vector_type(16))) __bf16 v16bf;
typedef __attribute__((ext_vector_type(8)))  __bf16 v8bf;
typedef __attribute__((ext_vector_type(8)))  float  v8f;

#define B_  2048
#define S_  23
#define D_  768
#define H_  8
#define HD_ 96
#define L_  50
#define M_  (B_ * S_)   // 47104 = 184 * 256

// ---------------------------------------------------------------------------
// f32 -> bf16 conversion (weights)
// ---------------------------------------------------------------------------
__global__ void cvt_bf16_kernel(const float* __restrict__ src,
                                __bf16* __restrict__ dst, int n) {
  int i = blockIdx.x * blockDim.x + threadIdx.x;
  if (i < n) dst[i] = (__bf16)src[i];
}

// ---------------------------------------------------------------------------
// base = exist ? cls : missing_table   (stored bf16 for WMMA GEMM)
// ---------------------------------------------------------------------------
__global__ void build_base_kernel(const float* __restrict__ cls,
                                  const int* __restrict__ exist,
                                  const float* __restrict__ miss,
                                  __bf16* __restrict__ base, int total) {
  int i = blockIdx.x * blockDim.x + threadIdx.x;
  if (i >= total) return;
  int d  = i % D_;
  int bs = i / D_;
  int s  = bs % S_;
  float v = exist[bs] ? cls[i] : miss[s * D_ + d];
  base[i] = (__bf16)v;
}

// ---------------------------------------------------------------------------
// out[M,N] = A[M,K] @ W[N,K]^T + bias[N], bf16 in, f32 accum, bf16 out.
// Wave tile: 32(M) x 64(N) = 8 x v_wmma_f32_16x16x32_bf16 per K-step of 32;
// the 4 B tiles are loaded once per K-step and reused by both A tiles.
// All 12 B128 loads for a K-step are issued before the 8 WMMAs; WMMAs are
// ordered (j,t) so both A tiles are consumed early, keeping all loads in one
// front clause with partial loadcnt waits instead of a full stall mid-step.
// Requires: M % 256 == 0, N % 64 == 0, K % 32 == 0 (true for all uses here).
// ---------------------------------------------------------------------------
__global__ void __launch_bounds__(256)
gemm_bf16_wmma_kernel(const __bf16* __restrict__ A,
                      const __bf16* __restrict__ W,
                      const float*  __restrict__ bias,
                      __bf16* __restrict__ out,
                      int M, int N, int K) {
  const int wave = threadIdx.x >> 5;
  const int lane = threadIdx.x & 31;
  const int ln16 = lane & 15;
  const int half = lane >> 4;          // 0: K {0-7,16-23}, 1: K {8-15,24-31}
  const int m0 = blockIdx.y * 256 + wave * 32;
  const int n0 = blockIdx.x * 64;

  v8f acc[2][4] = {};

  const __bf16* arow0 = A + (size_t)(m0 + ln16) * K;
  const __bf16* arow1 = A + (size_t)(m0 + 16 + ln16) * K;
  const __bf16* wrow[4];
#pragma unroll
  for (int j = 0; j < 4; ++j)
    wrow[j] = W + (size_t)(n0 + j * 16 + ln16) * K;

  const int hoff = half * 8;
  for (int kk = 0; kk < K; kk += 32) {
    // ---- issue ALL loads for this K-step first ----
    v8bf a0lo = *(const v8bf*)(arow0 + kk + hoff);
    v8bf a0hi = *(const v8bf*)(arow0 + kk + 16 + hoff);
    v8bf a1lo = *(const v8bf*)(arow1 + kk + hoff);
    v8bf a1hi = *(const v8bf*)(arow1 + kk + 16 + hoff);
    v8bf blo[4], bhi[4];
#pragma unroll
    for (int j = 0; j < 4; ++j) {
      blo[j] = *(const v8bf*)(wrow[j] + kk + hoff);
      bhi[j] = *(const v8bf*)(wrow[j] + kk + 16 + hoff);
    }
    v16bf a0 = __builtin_shufflevector(a0lo, a0hi,
        0, 1, 2, 3, 4, 5, 6, 7, 8, 9, 10, 11, 12, 13, 14, 15);
    v16bf a1 = __builtin_shufflevector(a1lo, a1hi,
        0, 1, 2, 3, 4, 5, 6, 7, 8, 9, 10, 11, 12, 13, 14, 15);
    v16bf bv[4];
#pragma unroll
    for (int j = 0; j < 4; ++j)
      bv[j] = __builtin_shufflevector(blo[j], bhi[j],
          0, 1, 2, 3, 4, 5, 6, 7, 8, 9, 10, 11, 12, 13, 14, 15);

    // ---- 8 WMMAs, interleaved so both A tiles are live from the start ----
#pragma unroll
    for (int j = 0; j < 4; ++j) {
      acc[0][j] = __builtin_amdgcn_wmma_f32_16x16x32_bf16(
          false, a0, false, bv[j], (short)0, acc[0][j], false, false);
      acc[1][j] = __builtin_amdgcn_wmma_f32_16x16x32_bf16(
          false, a1, false, bv[j], (short)0, acc[1][j], false, false);
    }
  }

  // C/D layout: VGPR r, lanes 0-15 -> row r, lanes 16-31 -> row r+8
#pragma unroll
  for (int t = 0; t < 2; ++t) {
#pragma unroll
    for (int j = 0; j < 4; ++j) {
      int n = n0 + j * 16 + ln16;
      float bval = bias[n];
#pragma unroll
      for (int r = 0; r < 8; ++r) {
        int mo = m0 + t * 16 + half * 8 + r;
        out[(size_t)mo * N + n] = (__bf16)(acc[t][j][r] + bval);
      }
    }
  }
}

// ---------------------------------------------------------------------------
// Per-(b,h) attention over S=23 keys, masked to existing sections.
// qkv: [B,S,3D] bf16;  ctx: [B,S,D] bf16
// ---------------------------------------------------------------------------
__global__ void __launch_bounds__(128)
attn_kernel(const __bf16* __restrict__ qkv,
            const int* __restrict__ exist,
            __bf16* __restrict__ ctx) {
  const int b = blockIdx.x / H_;
  const int h = blockIdx.x % H_;

  __shared__ float qs[S_][HD_];
  __shared__ float ks[S_][HD_];
  __shared__ float vs[S_][HD_];
  __shared__ float sc[S_][S_ + 1];
  __shared__ int   ex[S_];

  const size_t boff = (size_t)b * S_ * 3 * D_;
  for (int idx = threadIdx.x; idx < S_ * HD_; idx += 128) {
    int s = idx / HD_, d = idx % HD_;
    size_t o = boff + (size_t)s * 3 * D_ + h * HD_ + d;
    qs[s][d] = (float)qkv[o];
    ks[s][d] = (float)qkv[o + D_];
    vs[s][d] = (float)qkv[o + 2 * D_];
  }
  if (threadIdx.x < S_) ex[threadIdx.x] = exist[b * S_ + threadIdx.x];
  __syncthreads();

  const float scale = 0.10206207261596575f;  // 1/sqrt(96)
  for (int p = threadIdx.x; p < S_ * S_; p += 128) {
    int i = p / S_, j = p % S_;
    float dot = 0.f;
#pragma unroll 8
    for (int d = 0; d < HD_; ++d) dot += qs[i][d] * ks[j][d];
    sc[i][j] = ex[j] ? dot * scale : -1e9f;
  }
  __syncthreads();

  if (threadIdx.x < S_) {
    int i = threadIdx.x;
    float mx = -3.4e38f;
    for (int j = 0; j < S_; ++j) mx = fmaxf(mx, sc[i][j]);
    float sum = 0.f;
    for (int j = 0; j < S_; ++j) { float e = __expf(sc[i][j] - mx); sc[i][j] = e; sum += e; }
    float inv = 1.f / sum;
    for (int j = 0; j < S_; ++j) sc[i][j] *= inv;
  }
  __syncthreads();

  for (int idx = threadIdx.x; idx < S_ * HD_; idx += 128) {
    int i = idx / HD_, d = idx % HD_;
    float a = 0.f;
#pragma unroll
    for (int j = 0; j < S_; ++j) a += sc[i][j] * vs[j][d];
    ctx[((size_t)b * S_ + i) * D_ + h * HD_ + d] = (__bf16)a;
  }
}

// ---------------------------------------------------------------------------
// section_embeddings select + mean over S -> doc[B,D] f32
// keep = exist | !any_exist ; base recomputed in f32 for accuracy
// ---------------------------------------------------------------------------
__global__ void __launch_bounds__(256)
combine_mean_kernel(const float* __restrict__ cls,
                    const int* __restrict__ exist,
                    const float* __restrict__ miss,
                    const __bf16* __restrict__ attn_out,
                    float* __restrict__ doc) {
  const int b = blockIdx.x;
  __shared__ int anyex;
  if (threadIdx.x == 0) {
    int a = 0;
    for (int s = 0; s < S_; ++s) a |= exist[b * S_ + s];
    anyex = a;
  }
  __syncthreads();
  const int any = anyex;
  for (int d = threadIdx.x; d < D_; d += 256) {
    float sum = 0.f;
    for (int s = 0; s < S_; ++s) {
      int e = exist[b * S_ + s];
      size_t o = ((size_t)b * S_ + s) * D_ + d;
      float basev = e ? cls[o] : miss[s * D_ + d];
      float val = (e || !any) ? basev : (float)attn_out[o];
      sum += val;
    }
    doc[(size_t)b * D_ + d] = sum * (1.0f / S_);
  }
}

// ---------------------------------------------------------------------------
// logits[B,L] = doc @ pred_w^T + pred_b
// ---------------------------------------------------------------------------
__global__ void __launch_bounds__(64)
pred_kernel(const float* __restrict__ doc,
            const float* __restrict__ pw,
            const float* __restrict__ pb,
            float* __restrict__ out) {
  const int b = blockIdx.x;
  __shared__ float drow[D_];
  for (int d = threadIdx.x; d < D_; d += 64)
    drow[d] = doc[(size_t)b * D_ + d];
  __syncthreads();
  int l = threadIdx.x;
  if (l < L_) {
    float acc = pb[l];
#pragma unroll 8
    for (int d = 0; d < D_; ++d) acc += drow[d] * pw[l * D_ + d];
    out[b * L_ + l] = acc;
  }
}

// ---------------------------------------------------------------------------
extern "C" void kernel_launch(void* const* d_in, const int* in_sizes, int n_in,
                              void* d_out, int out_size, void* d_ws, size_t ws_size,
                              hipStream_t stream) {
  const float* cls    = (const float*)d_in[0];   // [B,S,D]
  const int*   exist  = (const int*)  d_in[1];   // [B,S]
  const float* miss   = (const float*)d_in[2];   // [S,D]
  const float* w_in   = (const float*)d_in[3];   // [3D,D]
  const float* b_in   = (const float*)d_in[4];   // [3D]
  const float* w_out  = (const float*)d_in[5];   // [D,D]
  const float* b_out  = (const float*)d_in[6];   // [D]
  const float* pw     = (const float*)d_in[7];   // [L,D]
  const float* pb     = (const float*)d_in[8];   // [L]
  float* logits       = (float*)d_out;           // [B,L]

  // Workspace carve-up (all 256B aligned)
  auto align256 = [](size_t x) { return (x + 255) & ~(size_t)255; };
  unsigned char* ws = (unsigned char*)d_ws;
  size_t off = 0;
  __bf16* base_bf   = (__bf16*)(ws + off); off += align256((size_t)M_ * D_ * 2);
  __bf16* qkv_bf    = (__bf16*)(ws + off); off += align256((size_t)M_ * 3 * D_ * 2);
  __bf16* ctx_bf    = (__bf16*)(ws + off); off += align256((size_t)M_ * D_ * 2);
  __bf16* attnout_bf= (__bf16*)(ws + off); off += align256((size_t)M_ * D_ * 2);
  float*  doc       = (float*) (ws + off); off += align256((size_t)B_ * D_ * 4);
  __bf16* w_in_bf   = (__bf16*)(ws + off); off += align256((size_t)3 * D_ * D_ * 2);
  __bf16* w_out_bf  = (__bf16*)(ws + off); off += align256((size_t)D_ * D_ * 2);
  (void)ws_size; (void)in_sizes; (void)n_in; (void)out_size;

  // 1) weights -> bf16
  {
    int n1 = 3 * D_ * D_;
    cvt_bf16_kernel<<<(n1 + 255) / 256, 256, 0, stream>>>(w_in, w_in_bf, n1);
    int n2 = D_ * D_;
    cvt_bf16_kernel<<<(n2 + 255) / 256, 256, 0, stream>>>(w_out, w_out_bf, n2);
  }

  // 2) base = select(exist, cls, missing) -> bf16
  {
    int total = M_ * D_;
    build_base_kernel<<<(total + 255) / 256, 256, 0, stream>>>(
        cls, exist, miss, base_bf, total);
  }

  // 3) QKV projection: [47104,768] @ [2304,768]^T + bias  (WMMA bf16)
  gemm_bf16_wmma_kernel<<<dim3(3 * D_ / 64, M_ / 256), 256, 0, stream>>>(
      base_bf, w_in_bf, b_in, qkv_bf, M_, 3 * D_, D_);

  // 4) masked multi-head attention per (b,h)
  attn_kernel<<<B_ * H_, 128, 0, stream>>>(qkv_bf, exist, ctx_bf);

  // 5) output projection: [47104,768] @ [768,768]^T + bias  (WMMA bf16)
  gemm_bf16_wmma_kernel<<<dim3(D_ / 64, M_ / 256), 256, 0, stream>>>(
      ctx_bf, w_out_bf, b_out, attnout_bf, M_, D_, D_);

  // 6) select kept sections + mean over S
  combine_mean_kernel<<<B_, 256, 0, stream>>>(cls, exist, miss, attnout_bf, doc);

  // 7) classifier head
  pred_kernel<<<B_, 64, 0, stream>>>(doc, pw, pb, logits);
}
